// SelfAttention_13752485282345
// MI455X (gfx1250) — matrix operand order repaired and checked by reading, not compiled, not run on hardware
//
#include <hip/hip_runtime.h>
#include <hip/hip_bf16.h>

typedef __attribute__((ext_vector_type(16))) __bf16 v16bf;
typedef __attribute__((ext_vector_type(8)))  __bf16 v8bf;
typedef __attribute__((ext_vector_type(8)))  float  v8f;

#define D_  1024
#define S_  2048
#define B_  4
#define H_  16
#define HD_ 64
#define M_  (B_ * S_)   // 8192 tokens

// ---------- helpers ----------
__device__ __forceinline__ v8bf ld8(const __bf16* p) { return *(const v8bf*)p; }

__device__ __forceinline__ v16bf pack16(v8bf lo, v8bf hi) {
  v16bf r;
#pragma unroll
  for (int e = 0; e < 8; ++e) { r[e] = lo[e]; r[e + 8] = hi[e]; }
  return r;
}

__device__ __forceinline__ v8f zero8() {
  v8f z;
#pragma unroll
  for (int i = 0; i < 8; ++i) z[i] = 0.0f;
  return z;
}

// low 32 bits of a generic pointer into LDS == LDS byte offset (aperture scheme)
__device__ __forceinline__ unsigned lds_off(const void* p) {
  return (unsigned)(unsigned long long)p;
}

// Async DMA: 32B per thread for A-slice and B-slice (2x b128 each; the
// instruction IOFFSET applies to both LDS and global addresses).
__device__ __forceinline__ void stage_async(unsigned long long gA, unsigned long long gB,
                                            unsigned lA, unsigned lB) {
  asm volatile(
      "global_load_async_to_lds_b128 %0, %2, off\n\t"
      "global_load_async_to_lds_b128 %0, %2, off offset:16\n\t"
      "global_load_async_to_lds_b128 %1, %3, off\n\t"
      "global_load_async_to_lds_b128 %1, %3, off offset:16"
      :: "v"(lA), "v"(lB), "v"(gA), "v"(gB)
      : "memory");
}

// ---------- f32 -> bf16 convert ----------
__global__ void __launch_bounds__(256) cvt_bf16_kernel(const float* __restrict__ in,
                                                       __bf16* __restrict__ out, int n) {
  int i = blockIdx.x * 256 + threadIdx.x;
  if (i < n) out[i] = (__bf16)in[i];
}

// ---------- 128x128 tile GEMM core (bf16 A,B -> f32 acc), K = D_ ----------
// 8 waves (2x4); each wave computes 64x32 as 4(M) x 2(N) wmma tiles.
// Double-buffered LDS fed by async global->LDS DMA (ASYNCcnt).
__device__ __forceinline__ void gemm_core_128x128(const __bf16* __restrict__ A,
                                                  const __bf16* __restrict__ Bm,
                                                  int blkM, int blkN,
                                                  __bf16* As, __bf16* Bs,  // [2][4096] each
                                                  v8f acc[4][2]) {
  const int tid  = threadIdx.x;
  const int wave = tid >> 5;
  const int lane = tid & 31;
  const int lo = lane & 15, hi = lane >> 4;
  const int wM = wave & 1, wN = wave >> 1;

#pragma unroll
  for (int mt = 0; mt < 4; ++mt)
#pragma unroll
    for (int nt = 0; nt < 2; ++nt) acc[mt][nt] = zero8();

  const int aRow = tid >> 1, aCol = (tid & 1) * 16;   // 128 rows x 32 cols
  const int bRow = tid >> 3, bCol = (tid & 7) * 16;   // 32 rows x 128 cols
  const long aBase = (long)(blkM * 128 + aRow) * D_ + aCol;
  const long bBase = (long)bRow * D_ + (long)blkN * 128 + bCol;

  const unsigned lA0 = lds_off(&As[aRow * 32 + aCol]);
  const unsigned lB0 = lds_off(&Bs[bRow * 128 + bCol]);

  // prologue: stage K-chunk 0 into buffer 0
  stage_async((unsigned long long)(A + aBase),
              (unsigned long long)(Bm + bBase),
              lA0, lB0);

  for (int ks = 0; ks < 32; ++ks) {
    const int cur = ks & 1;
    const int nxt = cur ^ 1;
    if (ks + 1 < 32) {  // stage next chunk into the other buffer
      stage_async((unsigned long long)(A + aBase + (ks + 1) * 32),
                  (unsigned long long)(Bm + bBase + (long)(ks + 1) * 32 * D_),
                  lA0 + nxt * 8192, lB0 + nxt * 8192);
      asm volatile("s_wait_asynccnt 0x4" ::: "memory");  // in-order: chunk ks landed
    } else {
      asm volatile("s_wait_asynccnt 0x0" ::: "memory");
    }
    __syncthreads();

    const __bf16* Ac = As + cur * 4096;
    const __bf16* Bc = Bs + cur * 4096;
    v16bf af[4], bfr[2];
#pragma unroll
    for (int mt = 0; mt < 4; ++mt) {
      const __bf16* p = &Ac[(wM * 64 + mt * 16 + lo) * 32 + hi * 8];
      af[mt] = pack16(ld8(p), ld8(p + 16));
    }
#pragma unroll
    for (int nt = 0; nt < 2; ++nt) {
      const __bf16* p = &Bc[wN * 32 + nt * 16 + lo];
      for (int e = 0; e < 8; ++e) { bfr[nt][e] = p[(hi * 8 + e) * 128]; bfr[nt][e + 8] = p[(16 + hi * 8 + e) * 128]; }
    }
#pragma unroll
    for (int mt = 0; mt < 4; ++mt)
#pragma unroll
      for (int nt = 0; nt < 2; ++nt)
        acc[mt][nt] = __builtin_amdgcn_wmma_f32_16x16x32_bf16(
            false, af[mt], false, bfr[nt], (short)0, acc[mt][nt], false, false);
    __syncthreads();  // reads done before buffer 'cur' is overwritten at ks+2
  }
}

// ---------- fused QKV projection: z selects Q/K/V ----------
__global__ void __launch_bounds__(256) qkv_gemm_kernel(
    const __bf16* __restrict__ xbf,
    const __bf16* __restrict__ Wq, const __bf16* __restrict__ Wk, const __bf16* __restrict__ Wv,
    const float* __restrict__ bq, const float* __restrict__ bk, const float* __restrict__ bv,
    __bf16* __restrict__ Qbf, __bf16* __restrict__ Ktr, __bf16* __restrict__ Vbf) {
  __shared__ __bf16 As[2][128 * 32];
  __shared__ __bf16 Bs[2][32 * 128];
  const int which = blockIdx.z;
  const __bf16* Bm  = (which == 0) ? Wq : (which == 1) ? Wk : Wv;
  const float*  bia = (which == 0) ? bq : (which == 1) ? bk : bv;

  v8f acc[4][2];
  gemm_core_128x128(xbf, Bm, blockIdx.x, blockIdx.y, &As[0][0], &Bs[0][0], acc);

  const int lane = threadIdx.x & 31, wave = threadIdx.x >> 5;
  const int lo = lane & 15, hi = lane >> 4, wM = wave & 1, wN = wave >> 1;
#pragma unroll
  for (int mt = 0; mt < 4; ++mt)
#pragma unroll
    for (int nt = 0; nt < 2; ++nt)
#pragma unroll
      for (int r = 0; r < 8; ++r) {
        int row = blockIdx.x * 128 + wM * 64 + mt * 16 + r + 8 * hi;
        int col = blockIdx.y * 128 + wN * 32 + nt * 16 + lo;
        float v = acc[mt][nt][r] + bia[col];
        int b = row >> 11, s = row & 2047;
        int h = col >> 6,  d = col & 63;
        long hb = (long)b * H_ + h;
        if (which == 0)       Qbf[((hb * S_ + s) << 6)  + d] = (__bf16)(v * 0.125f); // fold 1/sqrt(hd)
        else if (which == 1)  Ktr[((hb * HD_ + d) << 11) + s] = (__bf16)v;           // transposed K
        else                  Vbf[((hb * S_ + s) << 6)  + d] = (__bf16)v;
      }
}

// ---------- O projection + bias + residual ----------
__global__ void __launch_bounds__(256) oproj_gemm_kernel(
    const __bf16* __restrict__ ctxbf, const __bf16* __restrict__ Wo,
    const float* __restrict__ bo, const float* __restrict__ xres,
    float* __restrict__ yres) {
  __shared__ __bf16 As[2][128 * 32];
  __shared__ __bf16 Bs[2][32 * 128];
  v8f acc[4][2];
  gemm_core_128x128(ctxbf, Wo, blockIdx.x, blockIdx.y, &As[0][0], &Bs[0][0], acc);

  const int lane = threadIdx.x & 31, wave = threadIdx.x >> 5;
  const int lo = lane & 15, hi = lane >> 4, wM = wave & 1, wN = wave >> 1;
#pragma unroll
  for (int mt = 0; mt < 4; ++mt)
#pragma unroll
    for (int nt = 0; nt < 2; ++nt)
#pragma unroll
      for (int r = 0; r < 8; ++r) {
        int row = blockIdx.x * 128 + wM * 64 + mt * 16 + r + 8 * hi;
        int col = blockIdx.y * 128 + wN * 32 + nt * 16 + lo;
        long idx = (long)row * D_ + col;
        yres[idx] = acc[mt][nt][r] + bo[col] + xres[idx];
      }
}

// ---------- causal flash attention: 1 WG = 128 queries of one (b,h) ----------
__global__ void __launch_bounds__(256) attn_kernel(
    const __bf16* __restrict__ Qbf, const __bf16* __restrict__ Ktr,
    const __bf16* __restrict__ Vbf, __bf16* __restrict__ ctx) {
  __shared__ __bf16 Ps[8][16][32];  // per-wave P transpose staging
  const int b = blockIdx.z, h = blockIdx.y;
  const int wave = threadIdx.x >> 5, lane = threadIdx.x & 31;
  const int lo = lane & 15, hi = lane >> 4;
  const int qb = blockIdx.x * 128 + wave * 16;

  const long headQV = ((long)b * H_ + h) * S_;   // rows of [.., 64]
  const long headK  = ((long)b * H_ + h) * HD_;  // rows of [.., 2048]

  // resident Q A-fragments (hd = 64 -> 2 K-chunks of 32)
  const __bf16* qrow = Qbf + (headQV + qb + lo) * HD_;
  v16bf aQ[2];
#pragma unroll
  for (int c = 0; c < 2; ++c)
    aQ[c] = pack16(ld8(qrow + c * 32 + hi * 8), ld8(qrow + c * 32 + hi * 8 + 16));

  v8f o[4];
  float m[8], l[8];
#pragma unroll
  for (int n = 0; n < 4; ++n) o[n] = zero8();
#pragma unroll
  for (int r = 0; r < 8; ++r) { m[r] = -__builtin_inff(); l[r] = 0.0f; }

  for (int kb = 0; kb < qb + 16; kb += 32) {   // causal key blocks of 32
    // S = Q * K^T   (2 key tiles x 2 hd chunks)
    v8f st[2];
#pragma unroll
    for (int t = 0; t < 2; ++t) {
      const __bf16* kc0 = Ktr + headK * S_ + kb + t * 16 + lo;
      v16bf bK0;
      for (int e = 0; e < 8; ++e) { bK0[e] = kc0[(long)(hi * 8 + e) * S_]; bK0[e + 8] = kc0[(long)(16 + hi * 8 + e) * S_]; }
      st[t] = __builtin_amdgcn_wmma_f32_16x16x32_bf16(
          false, aQ[0], false, bK0, (short)0, zero8(), false, false);
      const __bf16* kc1 = Ktr + (headK + 32) * S_ + kb + t * 16 + lo;
      v16bf bK1;
      for (int e = 0; e < 8; ++e) { bK1[e] = kc1[(long)(hi * 8 + e) * S_]; bK1[e + 8] = kc1[(long)(16 + hi * 8 + e) * S_]; }
      st[t] = __builtin_amdgcn_wmma_f32_16x16x32_bf16(
          false, aQ[1], false, bK1, (short)0, st[t], false, false);
    }
    // causal mask (C layout: row = r + 8*hi, col = lo)
#pragma unroll
    for (int t = 0; t < 2; ++t)
#pragma unroll
      for (int r = 0; r < 8; ++r)
        if (kb + t * 16 + lo > qb + r + 8 * hi) st[t][r] = -__builtin_inff();

    // row max across 16 lanes of each half
    float rmax[8];
#pragma unroll
    for (int r = 0; r < 8; ++r) rmax[r] = fmaxf(st[0][r], st[1][r]);
#pragma unroll
    for (int msk = 1; msk < 16; msk <<= 1)
#pragma unroll
      for (int r = 0; r < 8; ++r) rmax[r] = fmaxf(rmax[r], __shfl_xor(rmax[r], msk, 32));

    float corr[8];
#pragma unroll
    for (int r = 0; r < 8; ++r) {
      float mn = fmaxf(m[r], rmax[r]);
      corr[r] = __expf(m[r] - mn);
      m[r] = mn;
      l[r] *= corr[r];
    }
#pragma unroll
    for (int n = 0; n < 4; ++n)
#pragma unroll
      for (int r = 0; r < 8; ++r) o[n][r] *= corr[r];

    float rs[8];
#pragma unroll
    for (int r = 0; r < 8; ++r) rs[r] = 0.0f;
#pragma unroll
    for (int t = 0; t < 2; ++t)
#pragma unroll
      for (int r = 0; r < 8; ++r) {
        float p = __expf(st[t][r] - m[r]);
        st[t][r] = p;
        rs[r] += p;
      }
#pragma unroll
    for (int msk = 1; msk < 16; msk <<= 1)
#pragma unroll
      for (int r = 0; r < 8; ++r) rs[r] += __shfl_xor(rs[r], msk, 32);
#pragma unroll
    for (int r = 0; r < 8; ++r) l[r] += rs[r];

    // transpose P (C layout -> A layout) through per-wave LDS tile
#pragma unroll
    for (int t = 0; t < 2; ++t)
#pragma unroll
      for (int r = 0; r < 8; ++r)
        Ps[wave][r + 8 * hi][t * 16 + lo] = (__bf16)st[t][r];
    v16bf aP = pack16(ld8(&Ps[wave][lo][hi * 8]), ld8(&Ps[wave][lo][hi * 8 + 16]));

    // O += P * V
#pragma unroll
    for (int n = 0; n < 4; ++n) {
      const __bf16* vrow = Vbf + (headQV + kb) * HD_ + n * 16 + lo;
      v16bf bV;
      for (int e = 0; e < 8; ++e) { bV[e] = vrow[(hi * 8 + e) * HD_]; bV[e + 8] = vrow[(16 + hi * 8 + e) * HD_]; }
      o[n] = __builtin_amdgcn_wmma_f32_16x16x32_bf16(
          false, aP, false, bV, (short)0, o[n], false, false);
    }
  }

  // normalize + store ctx in [B,S,D] bf16 (heads re-interleaved)
#pragma unroll
  for (int n = 0; n < 4; ++n)
#pragma unroll
    for (int r = 0; r < 8; ++r) {
      float val = o[n][r] / l[r];
      long row = (long)b * S_ + qb + r + 8 * hi;
      ctx[row * D_ + h * HD_ + n * 16 + lo] = (__bf16)val;
    }
}

// ---------- LayerNorm over rows of 1024 ----------
__global__ void __launch_bounds__(256) layernorm_kernel(
    const float* __restrict__ y, const float* __restrict__ gamma,
    const float* __restrict__ beta, float* __restrict__ out) {
  const long row = blockIdx.x;
  const float* yr = y + row * D_;
  float v[4], s = 0.0f, sq = 0.0f;
#pragma unroll
  for (int i = 0; i < 4; ++i) {
    v[i] = yr[threadIdx.x + 256 * i];
    s += v[i];
    sq += v[i] * v[i];
  }
#pragma unroll
  for (int msk = 1; msk < 32; msk <<= 1) {
    s  += __shfl_xor(s,  msk, 32);
    sq += __shfl_xor(sq, msk, 32);
  }
  __shared__ float rs[8], rq[8];
  const int wave = threadIdx.x >> 5, lane = threadIdx.x & 31;
  if (lane == 0) { rs[wave] = s; rq[wave] = sq; }
  __syncthreads();
  s = 0.0f; sq = 0.0f;
#pragma unroll
  for (int w = 0; w < 8; ++w) { s += rs[w]; sq += rq[w]; }
  float mu   = s * (1.0f / D_);
  float var  = sq * (1.0f / D_) - mu * mu;
  float rstd = rsqrtf(var + 1e-5f);
#pragma unroll
  for (int i = 0; i < 4; ++i) {
    int col = threadIdx.x + 256 * i;
    out[row * D_ + col] = (v[i] - mu) * rstd * gamma[col] + beta[col];
  }
}

// ---------- launch ----------
extern "C" void kernel_launch(void* const* d_in, const int* in_sizes, int n_in,
                              void* d_out, int out_size, void* d_ws, size_t ws_size,
                              hipStream_t stream) {
  const float* x     = (const float*)d_in[0];
  const float* Wq    = (const float*)d_in[1];
  const float* bq    = (const float*)d_in[2];
  const float* Wk    = (const float*)d_in[3];
  const float* bk    = (const float*)d_in[4];
  const float* Wv    = (const float*)d_in[5];
  const float* bv    = (const float*)d_in[6];
  const float* Wo    = (const float*)d_in[7];
  const float* bo    = (const float*)d_in[8];
  const float* gamma = (const float*)d_in[9];
  const float* beta  = (const float*)d_in[10];
  float* out = (float*)d_out;

  char* ws = (char*)d_ws;
  size_t off = 0;
  auto alloc = [&](size_t bytes) -> void* {
    void* p = ws + off;
    off += (bytes + 255) & ~(size_t)255;
    return p;
  };
  __bf16* xbf   = (__bf16*)alloc((size_t)M_ * D_ * 2);   // 16 MB
  __bf16* Wqb   = (__bf16*)alloc((size_t)D_ * D_ * 2);   //  2 MB
  __bf16* Wkb   = (__bf16*)alloc((size_t)D_ * D_ * 2);
  __bf16* Wvb   = (__bf16*)alloc((size_t)D_ * D_ * 2);
  __bf16* Wob   = (__bf16*)alloc((size_t)D_ * D_ * 2);
  __bf16* Qbf   = (__bf16*)alloc((size_t)M_ * D_ * 2);   // 16 MB
  __bf16* Ktr   = (__bf16*)alloc((size_t)M_ * D_ * 2);   // 16 MB
  __bf16* Vbf   = (__bf16*)alloc((size_t)M_ * D_ * 2);   // 16 MB
  __bf16* ctxbf = (__bf16*)alloc((size_t)M_ * D_ * 2);   // 16 MB
  // yres (32 MB f32) aliases Qbf+Ktr, which are dead after attn_kernel
  float* yres = (float*)Qbf;

  cvt_bf16_kernel<<<(M_ * D_) / 256, 256, 0, stream>>>(x,  xbf, M_ * D_);
  cvt_bf16_kernel<<<(D_ * D_) / 256, 256, 0, stream>>>(Wq, Wqb, D_ * D_);
  cvt_bf16_kernel<<<(D_ * D_) / 256, 256, 0, stream>>>(Wk, Wkb, D_ * D_);
  cvt_bf16_kernel<<<(D_ * D_) / 256, 256, 0, stream>>>(Wv, Wvb, D_ * D_);
  cvt_bf16_kernel<<<(D_ * D_) / 256, 256, 0, stream>>>(Wo, Wob, D_ * D_);

  qkv_gemm_kernel<<<dim3(M_ / 128, D_ / 128, 3), 256, 0, stream>>>(
      xbf, Wqb, Wkb, Wvb, bq, bk, bv, Qbf, Ktr, Vbf);

  attn_kernel<<<dim3(S_ / 128, H_, B_), 256, 0, stream>>>(Qbf, Ktr, Vbf, ctxbf);

  oproj_gemm_kernel<<<dim3(M_ / 128, D_ / 128, 1), 256, 0, stream>>>(
      ctxbf, Wob, bo, x, yres);

  layernorm_kernel<<<M_, 256, 0, stream>>>(yres, gamma, beta, out);
}
